// VNCMDLayerFunctional_249108103963
// MI455X (gfx1250) — matrix operand order repaired and checked by reading, not compiled, not run on hardware
//
#include <hip/hip_runtime.h>

#define B_      8
#define N_      16
#define T_      32768
#define ROWS    (B_ * N_)
#define FS_     2048.0f
#define SW_     1e-4f
#define TWO_PI  6.28318530717958647692f
#define K2PI    (TWO_PI / FS_)
#define THREADS 256
#define VEC     4
#define CHUNK   (THREADS * VEC)

typedef __attribute__((ext_vector_type(16))) _Float16 v16h;
typedef __attribute__((ext_vector_type(8)))  float    v8f;
typedef int v4i_ __attribute__((vector_size(4 * sizeof(int))));

// ---------------------------------------------------------------------------
// CDNA5 async global->LDS staging (double buffered), with safe fallback.
// Builtin signature (from compiler diagnostic): param0 = v4i addrspace(1)*,
// param1 = v4i addrspace(3)*, then imm offset, imm cpol.
// ---------------------------------------------------------------------------
#if defined(__gfx1250__) && __has_builtin(__builtin_amdgcn_global_load_async_to_lds_b128)
#define HAVE_ASYNC_LDS 1
#else
#define HAVE_ASYNC_LDS 0
#endif

#if HAVE_ASYNC_LDS
#define STAGE16(ldsp, gp)                                                      \
    __builtin_amdgcn_global_load_async_to_lds_b128(                            \
        (__attribute__((address_space(1))) v4i_*)(gp),                         \
        (__attribute__((address_space(3))) v4i_*)(ldsp), 0, 0)
#if __has_builtin(__builtin_amdgcn_s_wait_asynccnt)
#define ASYNC_WAIT(n) __builtin_amdgcn_s_wait_asynccnt(n)
#else
#define ASYNC_WAIT(n) asm volatile("s_wait_asynccnt %0" ::"n"(n) : "memory")
#endif
#else
#define STAGE16(ldsp, gp) (*(float4*)(ldsp) = *(const float4*)(gp))
#define ASYNC_WAIT(n)     do { } while (0)
#endif

// ---------------------------------------------------------------------------
// workspace layout (floats):
//  ws[0]      : sum(|eIF|) accumulator
//  ws[1]      : a (refined alpha)
//  ws[2]      : 1/a
//  ws[3]      : lr
//  ws[4+b]    : projection scale per batch (b < 8)
//  ws[32+row] : active flag per (b,n) row (row < 128), 1.0f / 0.0f
// ---------------------------------------------------------------------------

__global__ void k_zero(float* __restrict__ tail, int n, float* __restrict__ ws) {
    int i = blockIdx.x * blockDim.x + threadIdx.x;
    if (i < n)   tail[i] = 0.f;
    if (i < 256) ws[i]   = 0.f;
}

// --- per-row sum|eIF| and max|eIF| -> init_stat accumulator + active flags --
__global__ __launch_bounds__(THREADS)
void k_reduce_eif(const float* __restrict__ eIF, const int* __restrict__ mask,
                  float* __restrict__ ws) {
    int row = blockIdx.x, tid = threadIdx.x;
    const float* p = eIF + (size_t)row * T_;
    float s = 0.f, mx = 0.f;
    for (int t = tid; t < T_; t += THREADS) {
        float v = fabsf(p[t]);
        s += v;
        mx = fmaxf(mx, v);
    }
    __shared__ float ss[THREADS], sm[THREADS];
    ss[tid] = s; sm[tid] = mx;
    __syncthreads();
    for (int o = THREADS / 2; o; o >>= 1) {
        if (tid < o) { ss[tid] += ss[tid + o]; sm[tid] = fmaxf(sm[tid], sm[tid + o]); }
        __syncthreads();
    }
    if (tid == 0) {
        atomicAdd(&ws[0], ss[0]);
        ws[32 + row] = (sm[0] > 1e-6f && mask[row] != 0) ? 1.f : 0.f;
    }
}

// --- WMMA helpers: pack padded tiles for v_wmma_f32_16x16x32_f16 ------------
__device__ __forceinline__ v16h buildA16(const float* vals, int kbase, int kvalid, int lane) {
    v16h A;
    int m = lane & 15;
    int hiofs = (lane >= 16) ? 8 : 0;
#pragma unroll
    for (int e = 0; e < 16; e++) {
        int K = ((e < 8) ? e : (8 + e)) + hiofs;
        float x = 0.f;
        if (m == 0 && (kbase + K) < kvalid) x = vals[kbase + K];
        A[e] = (_Float16)x;
    }
    return A;
}
__device__ __forceinline__ v16h buildB16(const float* W, int ldn, int kbase, int kvalid,
                                         int cbase, int cvalid, int lane) {
    v16h Bv;
    int n = lane & 15;
    int kofs = (lane >= 16) ? 16 : 0;
#pragma unroll
    for (int e = 0; e < 16; e++) {
        int K = kbase + ((e < 8) ? e : (8 + e)) + (kofs >> 1);
        int c = cbase + n;
        float x = 0.f;
        if (K < kvalid && c < cvalid) x = W[K * ldn + c];
        Bv[e] = (_Float16)x;
    }
    return Bv;
}

// --- MLP (3->64->32->2, relu/relu/tanh) via WMMA; one wave32 ----------------
__global__ __launch_bounds__(32)
void k_mlp(const float* __restrict__ alpha, const float* __restrict__ beta,
           const float* __restrict__ W1, const float* __restrict__ b1,
           const float* __restrict__ W2, const float* __restrict__ b2,
           const float* __restrict__ W3, const float* __restrict__ b3,
           const int* __restrict__ iter, float* __restrict__ ws) {
    __shared__ float inp[4];
    __shared__ float h1[64];
    __shared__ float h2[32];
    int l = threadIdx.x;
    if (l == 0) {
        inp[0] = alpha[0];
        inp[1] = beta[0];
        inp[2] = ws[0] * (1.0f / ((float)B_ * (float)N_ * (float)T_)); // init_stat
        inp[3] = 0.f;
    }
    __syncthreads();

    // layer 1: [1x3] @ [3x64]
    v16h A1 = buildA16(inp, 0, 3, l);
#pragma unroll
    for (int j = 0; j < 4; j++) {
        v16h Bv = buildB16(W1, 64, 0, 3, 16 * j, 64, l);
        v8f c; for (int e = 0; e < 8; e++) c[e] = 0.f;
        c = __builtin_amdgcn_wmma_f32_16x16x32_f16(false, A1, false, Bv, (short)0, c, false, false);
        if (l < 16) h1[16 * j + l] = fmaxf(c[0] + b1[16 * j + l], 0.f);
    }
    __syncthreads();

    // layer 2: [1x64] @ [64x32], K accumulated over 2 chunks
#pragma unroll
    for (int j = 0; j < 2; j++) {
        v8f c; for (int e = 0; e < 8; e++) c[e] = 0.f;
#pragma unroll
        for (int kc = 0; kc < 2; kc++) {
            v16h A  = buildA16(h1, 32 * kc, 64, l);
            v16h Bv = buildB16(W2, 32, 32 * kc, 64, 16 * j, 32, l);
            c = __builtin_amdgcn_wmma_f32_16x16x32_f16(false, A, false, Bv, (short)0, c, false, false);
        }
        if (l < 16) h2[16 * j + l] = fmaxf(c[0] + b2[16 * j + l], 0.f);
    }
    __syncthreads();

    // layer 3: [1x32] @ [32x2]
    {
        v16h A  = buildA16(h2, 0, 32, l);
        v16h Bv = buildB16(W3, 2, 0, 32, 0, 2, l);
        v8f c; for (int e = 0; e < 8; e++) c[e] = 0.f;
        c = __builtin_amdgcn_wmma_f32_16x16x32_f16(false, A, false, Bv, (short)0, c, false, false);
        if (l == 0) {
            float ref0 = tanhf(c[0] + b3[0]);
            float al = alpha[0];
            float a = fmaxf(al + 0.3f * ref0 * al, 1e-6f);
            ws[1] = a;
            ws[2] = 1.f / a;
            ws[3] = 0.03f / (1.f + (float)iter[0] * 0.015f);
        }
    }
}

// --- per-batch ||v|| -> projection scale ------------------------------------
__global__ __launch_bounds__(THREADS)
void k_vnorm(const float* __restrict__ s, const float* __restrict__ sumx,
             const float* __restrict__ sumy, const float* __restrict__ lam,
             const float* __restrict__ var, float* __restrict__ ws) {
    int b = blockIdx.x, tid = threadIdx.x;
    float inv_a = ws[2];
    const float2* ps  = (const float2*)(s    + (size_t)b * 2 * T_);
    const float2* psx = (const float2*)(sumx + (size_t)b * 2 * T_);
    const float2* psy = (const float2*)(sumy + (size_t)b * 2 * T_);
    const float2* pl  = (const float2*)(lam  + (size_t)b * 2 * T_);
    float acc = 0.f;
    for (int t = tid; t < T_; t += THREADS) {
        float2 sv = ps[t], xv = psx[t], yv = psy[t], lv = pl[t];
        float re = sv.x - xv.x - yv.x - lv.x * inv_a;
        float im = sv.y - xv.y - yv.y - lv.y * inv_a;
        acc += re * re + im * im;
    }
    __shared__ float sb[THREADS];
    sb[tid] = acc;
    __syncthreads();
    for (int o = THREADS / 2; o; o >>= 1) {
        if (tid < o) sb[tid] += sb[tid + o];
        __syncthreads();
    }
    if (tid == 0) {
        float vr = var[0];
        float scale = 0.f;
        if (vr > 0.f) {
            float e   = sqrtf((float)T_ * vr);
            float nrm = sqrtf(sb[0]);
            scale = fminf(1.f, e / (nrm + 1e-12f));
        }
        ws[4 + b] = scale;
    }
}

// --- wave32 shuffle scan + small cross-wave combine -------------------------
__device__ __forceinline__ float wave_incl_scan(float v) {
#pragma unroll
    for (int d = 1; d < 32; d <<= 1) {
        float n = __shfl_up(v, d, 32);
        if ((int)(threadIdx.x & 31) >= d) v += n;
    }
    return v;
}
// Returns exclusive prefix; wtot[8] holds the chunk total afterwards.
// Caller must __syncthreads() before reusing wtot (done at end of chunk loop).
__device__ __forceinline__ float block_scan_excl(float v, float* wtot) {
    int tid = threadIdx.x, lane = tid & 31, w = tid >> 5;
    float incl = wave_incl_scan(v);
    if (lane == 31) wtot[w] = incl;
    __syncthreads();
    if (w == 0) {
        float t  = (lane < 8) ? wtot[lane] : 0.f;
        float ts = wave_incl_scan(t);
        if (lane < 8) wtot[lane] = ts - t;  // exclusive per-wave offset
        if (lane == 7) wtot[8] = ts;        // chunk total
    }
    __syncthreads();
    return (incl - v) + wtot[w];
}

// --- pass A: phase scan -> tot[b,t] += xm*cos + ym*sin (active modes) -------
__global__ __launch_bounds__(THREADS)
void k_passA(const float* __restrict__ eIF, const float* __restrict__ xm,
             const float* __restrict__ ym, const float* __restrict__ ws,
             float* __restrict__ tot) {
    int row = blockIdx.x, tid = threadIdx.x;
    int b = row >> 4;
    if (ws[32 + row] == 0.f) return;
    const float* pe  = eIF + (size_t)row * T_;
    const float* pxm = xm  + (size_t)row * 2 * T_;
    const float* pym = ym  + (size_t)row * 2 * T_;
    float* pt = tot + (size_t)b * 2 * T_;
    __shared__ float ebuf[2][CHUNK];
    __shared__ float wtot[16];
    STAGE16(&ebuf[0][tid * VEC], pe + tid * VEC);
    float carry = 0.f;
    int buf = 0;
    for (int c0 = 0; c0 < T_; c0 += CHUNK, buf ^= 1) {
        int t0 = c0 + tid * VEC;
        if (c0 + CHUNK < T_) {
            STAGE16(&ebuf[buf ^ 1][tid * VEC], pe + c0 + CHUNK + tid * VEC);
            ASYNC_WAIT(1);
        } else {
            ASYNC_WAIT(0);
        }
        float e0 = ebuf[buf][tid * VEC + 0], e1 = ebuf[buf][tid * VEC + 1];
        float e2 = ebuf[buf][tid * VEC + 2], e3 = ebuf[buf][tid * VEC + 3];
        float p1 = e0, p2 = p1 + e1, p3 = p2 + e2, p4 = p3 + e3;
        float excl  = block_scan_excl(p4, wtot);
        float total = wtot[8];
        float base  = carry + excl;
        float pref[4] = { p1, p2, p3, p4 };
        float4 m0 = *(const float4*)(pxm + 2 * t0);
        float4 m1 = *(const float4*)(pxm + 2 * t0 + 4);
        float4 n0 = *(const float4*)(pym + 2 * t0);
        float4 n1 = *(const float4*)(pym + 2 * t0 + 4);
        float xre[4] = { m0.x, m0.z, m1.x, m1.z }, xim[4] = { m0.y, m0.w, m1.y, m1.w };
        float yre[4] = { n0.x, n0.z, n1.x, n1.z }, yim[4] = { n0.y, n0.w, n1.y, n1.w };
#pragma unroll
        for (int i = 0; i < VEC; i++) {
            int t = t0 + i;
            float sn, cs;
            __sincosf(K2PI * (base + pref[i]), &sn, &cs);
            atomicAdd(&pt[2 * t],     xre[i] * cs + yre[i] * sn);
            atomicAdd(&pt[2 * t + 1], xim[i] * cs + yim[i] * sn);
        }
        carry += total;
        __syncthreads();
    }
}

// --- pass B: residual + gradients -> new_xm / new_ym ------------------------
__global__ __launch_bounds__(THREADS)
void k_passB(const float* __restrict__ s, const float* __restrict__ eIF,
             const float* __restrict__ xm, const float* __restrict__ ym,
             const float* __restrict__ sumx, const float* __restrict__ sumy,
             const float* __restrict__ lam, const float* __restrict__ tot,
             const float* __restrict__ ws,
             float* __restrict__ oxm, float* __restrict__ oym) {
    int row = blockIdx.x, tid = threadIdx.x;
    int b = row >> 4;
    const float* pxm = xm + (size_t)row * 2 * T_;
    const float* pym = ym + (size_t)row * 2 * T_;
    float* poxm = oxm + (size_t)row * 2 * T_;
    float* poym = oym + (size_t)row * 2 * T_;
    if (ws[32 + row] == 0.f) {   // inactive: passthrough
        for (int i = tid; i < 2 * T_ / 4; i += THREADS) {
            ((float4*)poxm)[i] = ((const float4*)pxm)[i];
            ((float4*)poym)[i] = ((const float4*)pym)[i];
        }
        return;
    }
    const float inv_a = ws[2], lr = ws[3], scale = ws[4 + b];
    const float* pe  = eIF  + (size_t)row * T_;
    const float* ps  = s    + (size_t)b * 2 * T_;
    const float* psx = sumx + (size_t)b * 2 * T_;
    const float* psy = sumy + (size_t)b * 2 * T_;
    const float* pl  = lam  + (size_t)b * 2 * T_;
    const float* pt  = tot  + (size_t)b * 2 * T_;
    __shared__ float ebuf[2][CHUNK];
    __shared__ float wtot[16];
    STAGE16(&ebuf[0][tid * VEC], pe + tid * VEC);
    float carry = 0.f;
    int buf = 0;
    for (int c0 = 0; c0 < T_; c0 += CHUNK, buf ^= 1) {
        int t0 = c0 + tid * VEC;
        if (c0 + CHUNK < T_) {
            STAGE16(&ebuf[buf ^ 1][tid * VEC], pe + c0 + CHUNK + tid * VEC);
            ASYNC_WAIT(1);
        } else {
            ASYNC_WAIT(0);
        }
        float e0 = ebuf[buf][tid * VEC + 0], e1 = ebuf[buf][tid * VEC + 1];
        float e2 = ebuf[buf][tid * VEC + 2], e3 = ebuf[buf][tid * VEC + 3];
        float p1 = e0, p2 = p1 + e1, p3 = p2 + e2, p4 = p3 + e3;
        float excl  = block_scan_excl(p4, wtot);
        float total = wtot[8];
        float base  = carry + excl;
        float pref[4] = { p1, p2, p3, p4 };
        // xm/ym covering t0-1 .. t0+4 (halo for second_diff_pad)
        float xre[6], xim[6], yre[6], yim[6];
        {
            int tm = (t0 == 0) ? 0 : (t0 - 1);
            float2 a = *(const float2*)(pxm + 2 * tm); xre[0] = a.x; xim[0] = a.y;
            float2 c = *(const float2*)(pym + 2 * tm); yre[0] = c.x; yim[0] = c.y;
            float4 m0 = *(const float4*)(pxm + 2 * t0);
            float4 m1 = *(const float4*)(pxm + 2 * t0 + 4);
            xre[1] = m0.x; xim[1] = m0.y; xre[2] = m0.z; xim[2] = m0.w;
            xre[3] = m1.x; xim[3] = m1.y; xre[4] = m1.z; xim[4] = m1.w;
            float4 n0 = *(const float4*)(pym + 2 * t0);
            float4 n1 = *(const float4*)(pym + 2 * t0 + 4);
            yre[1] = n0.x; yim[1] = n0.y; yre[2] = n0.z; yim[2] = n0.w;
            yre[3] = n1.x; yim[3] = n1.y; yre[4] = n1.z; yim[4] = n1.w;
            int tp = (t0 + 4 >= T_) ? (T_ - 1) : (t0 + 4);
            float2 bq = *(const float2*)(pxm + 2 * tp); xre[5] = bq.x; xim[5] = bq.y;
            float2 dq = *(const float2*)(pym + 2 * tp); yre[5] = dq.x; yim[5] = dq.y;
        }
#pragma unroll
        for (int i = 0; i < VEC; i++) {
            int t = t0 + i;
            float sn, cs;
            __sincosf(K2PI * (base + pref[i]), &sn, &cs);
            float2 sv  = *(const float2*)(ps  + 2 * t);
            float2 sxv = *(const float2*)(psx + 2 * t);
            float2 syv = *(const float2*)(psy + 2 * t);
            float2 lv  = *(const float2*)(pl  + 2 * t);
            float2 tv  = *(const float2*)(pt  + 2 * t);
            float la_re = lv.x * inv_a, la_im = lv.y * inv_a;
            float v_re = sv.x - sxv.x - syv.x - la_re;
            float v_im = sv.y - sxv.y - syv.y - la_im;
            float u_re = scale * v_re, u_im = scale * v_im;
            float w_re = sv.x - u_re - la_re;
            float w_im = sv.y - u_im - la_im;
            float txr = xre[i + 1] * cs, txi = xim[i + 1] * cs;
            float tyr = yre[i + 1] * sn, tyi = yim[i + 1] * sn;
            float res_re = w_re - (tv.x - txr - tyr);
            float res_im = w_im - (tv.y - txi - tyi);
            float sdpx_re = 0.f, sdpx_im = 0.f, sdpy_re = 0.f, sdpy_im = 0.f;
            if (t > 0 && t < T_ - 1) {
                sdpx_re = -SW_ * (xre[i + 2] - 2.f * xre[i + 1] + xre[i]);
                sdpx_im = -SW_ * (xim[i + 2] - 2.f * xim[i + 1] + xim[i]);
                sdpy_re = -SW_ * (yre[i + 2] - 2.f * yre[i + 1] + yre[i]);
                sdpy_im = -SW_ * (yim[i + 2] - 2.f * yim[i + 1] + yim[i]);
            }
            // rc = conj(res); xg = cos*rc + sdp(xm); yg = sin*rc + sdp(ym)
            float xg_re = cs * res_re + sdpx_re;
            float xg_im = -cs * res_im + sdpx_im;
            float yg_re = sn * res_re + sdpy_re;
            float yg_im = -sn * res_im + sdpy_im;
            float2 nxv; nxv.x = xre[i + 1] + lr * xg_re; nxv.y = xim[i + 1] + lr * xg_im;
            float2 nyv; nyv.x = yre[i + 1] + lr * yg_re; nyv.y = yim[i + 1] + lr * yg_im;
            *(float2*)(poxm + 2 * t) = nxv;
            *(float2*)(poym + 2 * t) = nyv;
        }
        carry += total;
        __syncthreads();
    }
}

// --- pass C: dIF -> new_eIF; scan new_eIF -> new_sum_x / new_sum_y ----------
__global__ __launch_bounds__(THREADS)
void k_passC(const float* __restrict__ eIF, const float* __restrict__ ws,
             const float* __restrict__ oxm, const float* __restrict__ oym,
             float* __restrict__ oeif, float* __restrict__ osx, float* __restrict__ osy) {
    int row = blockIdx.x, tid = threadIdx.x;
    int b = row >> 4;
    const float* pe = eIF + (size_t)row * T_;
    float* po = oeif + (size_t)row * T_;
    if (ws[32 + row] == 0.f) {   // inactive: eIF passthrough, no sum contribution
        for (int i = tid; i < T_ / 4; i += THREADS)
            ((float4*)po)[i] = ((const float4*)pe)[i];
        return;
    }
    const float lr = ws[3];
    const float* poxm = oxm + (size_t)row * 2 * T_;
    const float* poym = oym + (size_t)row * 2 * T_;
    float* psx = osx + (size_t)b * 2 * T_;
    float* psy = osy + (size_t)b * 2 * T_;
    __shared__ float ebuf[2][CHUNK];
    __shared__ float wtot[16];
    STAGE16(&ebuf[0][tid * VEC], pe + tid * VEC);
    float carry = 0.f;
    int buf = 0;
    for (int c0 = 0; c0 < T_; c0 += CHUNK, buf ^= 1) {
        int t0 = c0 + tid * VEC;
        if (c0 + CHUNK < T_) {
            STAGE16(&ebuf[buf ^ 1][tid * VEC], pe + c0 + CHUNK + tid * VEC);
            ASYNC_WAIT(1);
        } else {
            ASYNC_WAIT(0);
        }
        __syncthreads();  // cross-lane LDS reads below: all waves' tiles landed
        float ev[6];
        ev[0] = (tid > 0) ? ebuf[buf][tid * VEC - 1]
                          : pe[(t0 == 0) ? 0 : (t0 - 1)];
        ev[1] = ebuf[buf][tid * VEC + 0];
        ev[2] = ebuf[buf][tid * VEC + 1];
        ev[3] = ebuf[buf][tid * VEC + 2];
        ev[4] = ebuf[buf][tid * VEC + 3];
        ev[5] = (tid < THREADS - 1) ? ebuf[buf][tid * VEC + 4]
                                    : pe[(t0 + 4 >= T_) ? (T_ - 1) : (t0 + 4)];
        float xr[6], yr[6];
#pragma unroll
        for (int j = 0; j < 6; j++) {
            int t = t0 - 1 + j;
            t = (t < 0) ? 0 : ((t >= T_) ? (T_ - 1) : t);
            xr[j] = poxm[2 * t];
            yr[j] = poym[2 * t];
        }
        float ne[4];
#pragma unroll
        for (int i = 0; i < VEC; i++) {
            int t = t0 + i;
            float xc = xr[i + 1], yc = yr[i + 1];
            float xbar, ybar;
            if (t == 0)            { xbar = (xr[2] - xr[1]) * FS_;          ybar = (yr[2] - yr[1]) * FS_; }
            else if (t == T_ - 1)  { xbar = (xr[i + 1] - xr[i]) * FS_;      ybar = (yr[i + 1] - yr[i]) * FS_; }
            else                   { xbar = (xr[i + 2] - xr[i]) * (FS_ * 0.5f); ybar = (yr[i + 2] - yr[i]) * (FS_ * 0.5f); }
            float den  = xc * xc + yc * yc + 1e-12f;
            float sdpe = (t > 0 && t < T_ - 1) ? -SW_ * (ev[i + 2] - 2.f * ev[i + 1] + ev[i]) : 0.f;
            float dIF  = (xc * ybar - yc * xbar) / (TWO_PI * den) + sdpe;
            ne[i] = fminf(fmaxf(ev[i + 1] - lr * dIF, 2.0f), FS_ * 0.5f - 2.0f);
        }
        float4 o4; o4.x = ne[0]; o4.y = ne[1]; o4.z = ne[2]; o4.w = ne[3];
        *(float4*)(po + t0) = o4;
        float p1 = ne[0], p2 = p1 + ne[1], p3 = p2 + ne[2], p4 = p3 + ne[3];
        float excl  = block_scan_excl(p4, wtot);
        float total = wtot[8];
        float base  = carry + excl;
        float pref[4] = { p1, p2, p3, p4 };
#pragma unroll
        for (int i = 0; i < VEC; i++) {
            int t = t0 + i;
            float sn, cs;
            __sincosf(K2PI * (base + pref[i]), &sn, &cs);
            float2 xv = *(const float2*)(poxm + 2 * t);
            float2 yv = *(const float2*)(poym + 2 * t);
            atomicAdd(&psx[2 * t],     xv.x * cs);
            atomicAdd(&psx[2 * t + 1], xv.y * cs);
            atomicAdd(&psy[2 * t],     yv.x * sn);
            atomicAdd(&psy[2 * t + 1], yv.y * sn);
        }
        carry += total;
        __syncthreads();
    }
}

// --- new_lamuda --------------------------------------------------------------
__global__ void k_lam(const float* __restrict__ s, const float* __restrict__ sumx,
                      const float* __restrict__ sumy, const float* __restrict__ lam,
                      const float* __restrict__ osx, const float* __restrict__ osy,
                      const float* __restrict__ ws, float* __restrict__ olam) {
    int idx = blockIdx.x * blockDim.x + threadIdx.x;
    if (idx >= B_ * T_) return;
    int b = idx / T_;
    float a = ws[1], inv_a = ws[2], scale = ws[4 + b];
    float2 sv  = ((const float2*)s)[idx];
    float2 sxv = ((const float2*)sumx)[idx];
    float2 syv = ((const float2*)sumy)[idx];
    float2 lv  = ((const float2*)lam)[idx];
    float2 nx  = ((const float2*)osx)[idx];
    float2 ny  = ((const float2*)osy)[idx];
    float v_re = sv.x - sxv.x - syv.x - lv.x * inv_a;
    float v_im = sv.y - sxv.y - syv.y - lv.y * inv_a;
    float2 o;
    o.x = lv.x + a * (scale * v_re + nx.x + ny.x - sv.x);
    o.y = lv.y + a * (scale * v_im + nx.y + ny.y - sv.y);
    ((float2*)olam)[idx] = o;
}

// ---------------------------------------------------------------------------
extern "C" void kernel_launch(void* const* d_in, const int* in_sizes, int n_in,
                              void* d_out, int out_size, void* d_ws, size_t ws_size,
                              hipStream_t stream) {
    const float* s     = (const float*)d_in[0];
    const float* eIF   = (const float*)d_in[1];
    const float* xm    = (const float*)d_in[2];
    const float* ym    = (const float*)d_in[3];
    const float* sumx  = (const float*)d_in[4];
    const float* sumy  = (const float*)d_in[5];
    const float* lam   = (const float*)d_in[6];
    const float* alpha = (const float*)d_in[7];
    const float* beta  = (const float*)d_in[8];
    const float* W1    = (const float*)d_in[9];
    const float* b1    = (const float*)d_in[10];
    const float* W2    = (const float*)d_in[11];
    const float* b2    = (const float*)d_in[12];
    const float* W3    = (const float*)d_in[13];
    const float* b3    = (const float*)d_in[14];
    const int*   mask  = (const int*)d_in[15];
    const float* var   = (const float*)d_in[16];
    const int*   iter  = (const int*)d_in[17];

    float* out  = (float*)d_out;
    float* ws   = (float*)d_ws;
    float* o_eif = out;                                  // [B,N,T]   f32
    float* o_xm  = o_eif + (size_t)ROWS * T_;            // [B,N,T]   c64
    float* o_ym  = o_xm  + (size_t)ROWS * 2 * T_;        // [B,N,T]   c64
    float* o_sx  = o_ym  + (size_t)ROWS * 2 * T_;        // [B,T]     c64
    float* o_sy  = o_sx  + (size_t)B_ * 2 * T_;          // [B,T]     c64
    float* o_lam = o_sy  + (size_t)B_ * 2 * T_;          // [B,T]     c64
    float* tot   = o_lam;  // stage tot[b,t] in the lamuda slot; overwritten by k_lam

    int ntail = 3 * B_ * 2 * T_;  // zero sx, sy, lam(tot) regions
    k_zero<<<dim3((ntail + 255) / 256), dim3(256), 0, stream>>>(o_sx, ntail, ws);
    k_reduce_eif<<<dim3(ROWS), dim3(THREADS), 0, stream>>>(eIF, mask, ws);
    k_mlp<<<dim3(1), dim3(32), 0, stream>>>(alpha, beta, W1, b1, W2, b2, W3, b3, iter, ws);
    k_vnorm<<<dim3(B_), dim3(THREADS), 0, stream>>>(s, sumx, sumy, lam, var, ws);
    k_passA<<<dim3(ROWS), dim3(THREADS), 0, stream>>>(eIF, xm, ym, ws, tot);
    k_passB<<<dim3(ROWS), dim3(THREADS), 0, stream>>>(s, eIF, xm, ym, sumx, sumy, lam, tot, ws, o_xm, o_ym);
    k_passC<<<dim3(ROWS), dim3(THREADS), 0, stream>>>(eIF, ws, o_xm, o_ym, o_eif, o_sx, o_sy);
    k_lam<<<dim3((B_ * T_ + 255) / 256), dim3(256), 0, stream>>>(s, sumx, sumy, lam, o_sx, o_sy, ws, o_lam);
}
